// CNF_59133109731627
// MI455X (gfx1250) — matrix operand order repaired and checked by reading, not compiled, not run on hardware
//
#include <hip/hip_runtime.h>

// ---------------------------------------------------------------------------
// CNF ODE func for MI455X (gfx1250, wave32, WMMA).
// Bandwidth-bound: ~262 MiB mandatory HBM traffic -> ~11us floor @ 23.3 TB/s.
// Both GEMMs via v_wmma_f32_16x16x32_f16, weights register-resident, h repack
// via lane-half exchange (v_permlane16_swap_b32 if available, else DS shuffle),
// bias folded into the WMMA accumulator init.
// ---------------------------------------------------------------------------

typedef __attribute__((ext_vector_type(16))) _Float16 v16h;
typedef __attribute__((ext_vector_type(8)))  _Float16 h8v;
typedef __attribute__((ext_vector_type(2)))  __fp16   pk2h;  // cvt_pkrtz result
typedef __attribute__((ext_vector_type(8)))  float    v8f;
typedef __attribute__((ext_vector_type(4)))  float    fx4;

#define D_DIM   64
#define HID     512
#define WIDTH   64
#define BATCH_N 524288
#define BLK     4096          // WIDTH * D
#define P3N     12352         // 3*BLK + WIDTH
#define NTILES  (BATCH_N / 16)

// workspace byte offsets (all 16B aligned)
#define WS_P2   0             // 512 f32
#define WS_P3   2048          // 12352 f32
#define WS_WH   51456         // 4096 f16  : W  (width, d) row-major
#define WS_UTH  59648         // 4096 f16  : (U*sigmoid(G)/64)^T  (d, width) row-major
#define WS_WU   67840         // 64 f32    : sum_d W*U/64  per width
#define WS_BIAS 68096         // 64 f32

__device__ __forceinline__ float fast_tanh(float x) {
  float a = __builtin_fabsf(x);
  float e = __expf(-2.0f * a);
  float r = (1.0f - e) / (1.0f + e);
  return __builtin_copysignf(r, x);
}

// pack two f32 -> b32 holding an f16 pair (v_cvt_pkrtz_f16_f32)
__device__ __forceinline__ float packh2(float a, float b) {
  union { pk2h h; float f; } u;
  u.h = __builtin_amdgcn_cvt_pkrtz(a, b);
  return u.f;
}

__device__ __forceinline__ v8f wmma16(v16h a, v16h b, v8f c) {
  // v_wmma_f32_16x16x32_f16 (8-arg: neg_a, A, neg_b, B, c_mod, C, reuse_a, reuse_b)
  return __builtin_amdgcn_wmma_f32_16x16x32_f16(false, a, false, b, (short)0, c,
                                                false, false);
}

// Exchange lane halves between two b32 values:
//   out.x = lanes 0-15: a(own)        | lanes 16-31: b(from lane-16)
//   out.y = lanes 0-15: a(from l+16)  | lanes 16-31: b(own)
__device__ __forceinline__ void half_swap(float a, float b, float* lo, float* hi) {
#if __has_builtin(__builtin_amdgcn_permlane16_swap)
  union { float f; unsigned u; } ua, ub, r0, r1;
  ua.f = a; ub.f = b;
  auto s = __builtin_amdgcn_permlane16_swap(ua.u, ub.u, false, false);
  r0.u = s[0]; r1.u = s[1];
  *lo = r0.f; *hi = r1.f;
#else
  const int hf = (int)(threadIdx.x & 31) >> 4;
  float pa = __shfl_xor(a, 16, 32);
  float pb = __shfl_xor(b, 16, 32);
  *lo = hf ? pb : a;
  *hi = hf ? b : pa;
#endif
}

// A-matrix fragment (16x32 f16): per lane, 16 contiguous K values [16h,16h+16)
// of the lane's row -> two contiguous b128 loads.
__device__ __forceinline__ v16h load_a_frag(const _Float16* p) {
  h8v lo = *(const h8v*)p;
  h8v hi = *(const h8v*)(p + 8);
  v16h r;
#pragma unroll
  for (int i = 0; i < 8; ++i) { r[i] = lo[i]; r[8 + i] = hi[i]; }
  return r;
}

// B-matrix fragment (32x16 f16): per lane, 16 contiguous K values starting at
// 16*h of the lane's column -> four b128 f32 loads + cvt to f16.
__device__ __forceinline__ v16h load_b_frag_f32(const float* p) {
  fx4 a = *(const fx4*)p;
  fx4 b = *(const fx4*)(p + 4);
  fx4 c = *(const fx4*)(p + 8);
  fx4 d = *(const fx4*)(p + 12);
  v16h r;
#pragma unroll
  for (int i = 0; i < 4; ++i) {
    r[i]      = (_Float16)a[i];
    r[4 + i]  = (_Float16)b[i];
    r[8 + i]  = (_Float16)c[i];
    r[12 + i] = (_Float16)d[i];
  }
  return r;
}

// ------------------------- hypernetwork stage 1 ----------------------------
__global__ __launch_bounds__(512) void cnf_hyper1(
    const float* __restrict__ t,  const float* __restrict__ W1,
    const float* __restrict__ b1, const float* __restrict__ W2,
    const float* __restrict__ b2, float* __restrict__ p2out) {
  __shared__ float p1s[HID];
  const int j = threadIdx.x;
  const float tv = t[0];
  p1s[j] = fast_tanh(tv * W1[j] + b1[j]);
  __syncthreads();
  float acc = b2[j];
#pragma unroll 8
  for (int i = 0; i < HID; ++i) acc = fmaf(p1s[i], W2[i * HID + j], acc);
  p2out[j] = fast_tanh(acc);
}

// ------------------------- hypernetwork stage 2 ----------------------------
__global__ __launch_bounds__(128) void cnf_hyper2(
    const float* __restrict__ p2, const float* __restrict__ W3,
    const float* __restrict__ b3, float* __restrict__ p3) {
  __shared__ float p2s[HID];
  for (int i = threadIdx.x; i < HID; i += 128) p2s[i] = p2[i];
  __syncthreads();
  const int j = blockIdx.x * 128 + threadIdx.x;
  if (j >= P3N) return;
  float acc = b3[j];
#pragma unroll 8
  for (int i = 0; i < HID; ++i)
    acc = fmaf(p2s[i], W3[(size_t)i * P3N + j], acc);
  p3[j] = acc;
}

// -------- derive W(f16), (U*sigmoid(G)/64)^T(f16), wu, bias ----------------
__global__ __launch_bounds__(64) void cnf_derive(
    const float* __restrict__ p3, _Float16* __restrict__ Wh,
    _Float16* __restrict__ Uth, float* __restrict__ wu,
    float* __restrict__ bias) {
  __shared__ float red[64];
  const int w = blockIdx.x, d = threadIdx.x;
  const float Wv = p3[w * 64 + d];
  const float g  = p3[2 * BLK + w * 64 + d];
  const float Uv = p3[BLK + w * 64 + d] * (1.0f / (1.0f + __expf(-g)));
  const float Ue = Uv * (1.0f / 64.0f);            // fold 1/WIDTH into U
  Wh[w * 64 + d]  = (_Float16)Wv;
  Uth[d * 64 + w] = (_Float16)Ue;                  // U^T for GEMM2 A-frags
  red[d] = Wv * Ue;                                // wu with 1/WIDTH folded in
  __syncthreads();
  if (d == 0) {
    float s = 0.0f;
    for (int i = 0; i < 64; ++i) s += red[i];
    wu[w]   = s;
    bias[w] = p3[3 * BLK + w];
  }
}

// ------------------------------- main --------------------------------------
// Per wave: one 16-row batch tile per iteration.
//  GEMM1: h^T = W @ z^T        (M=w[4 tiles], N=batch16, K=d=64 -> 8 WMMA)
//  GEMM2: dz^T = U_eff^T @ h^T (M=d[4 tiles], N=batch16, K=w=64 -> 8 WMMA)
__global__ __launch_bounds__(256) void cnf_main(
    const float* __restrict__ z, const _Float16* __restrict__ Wh,
    const _Float16* __restrict__ Uth, const float* __restrict__ wu,
    const float* __restrict__ bias, float* __restrict__ dz,
    float* __restrict__ dlp) {
  const int lane = threadIdx.x & 31;
  const int wid  = (int)((blockIdx.x * blockDim.x + threadIdx.x) >> 5);
  const int nw   = (int)((gridDim.x * blockDim.x) >> 5);
  const int r    = lane & 15;   // fragment row / batch column
  const int hf   = lane >> 4;   // lane half

  // loop-invariant weight fragments (register resident)
  v16h Wf[4][2], Uf[4][2];
#pragma unroll
  for (int t = 0; t < 4; ++t)
#pragma unroll
    for (int s = 0; s < 2; ++s) {
      Wf[t][s] = load_a_frag(Wh  + (t * 16 + r) * 64 + s * 32 + 16 * hf);
      Uf[t][s] = load_a_frag(Uth + (t * 16 + r) * 64 + s * 32 + 16 * hf);
    }
  // per-lane bias / wu slices: (t,i) element has w = t*16 + i + 8*hf
  float br[4][8], wr[4][8];
#pragma unroll
  for (int t = 0; t < 4; ++t) {
    fx4 b0 = *(const fx4*)(bias + t * 16 + 8 * hf);
    fx4 b1 = *(const fx4*)(bias + t * 16 + 8 * hf + 4);
    fx4 w0 = *(const fx4*)(wu + t * 16 + 8 * hf);
    fx4 w1 = *(const fx4*)(wu + t * 16 + 8 * hf + 4);
#pragma unroll
    for (int i = 0; i < 4; ++i) {
      br[t][i] = b0[i]; br[t][4 + i] = b1[i];
      wr[t][i] = w0[i]; wr[t][4 + i] = w1[i];
    }
  }

  for (int tile = wid; tile < NTILES; tile += nw) {
    const int mb = tile * 16;
    const float* zr = z + (size_t)(mb + r) * 64;
    if (tile + nw < NTILES)  // global_prefetch_b8 next tile's rows
      __builtin_prefetch(z + (size_t)(mb + nw * 16 + r) * 64 + 16 * hf, 0, 3);

    // z B-fragments: 16 contiguous f32 per lane per K-step, cvt to f16
    v16h zf0 = load_b_frag_f32(zr + 16 * hf);
    v16h zf1 = load_b_frag_f32(zr + 32 + 16 * hf);

    // GEMM1 tile-by-tile: bias pre-loaded into the accumulator; tanh + trace
    // + f16-pair packing immediately so only one f32 accumulator stays live.
    // D element (lane, vgpr i) -> (batch = mb+r, w = t*16 + i + 8*hf)
    float hp[4][4];  // packed f16 pairs of h (b32 views)
    float g = 0.0f;
#pragma unroll
    for (int t = 0; t < 4; ++t) {
      v8f acc = {br[t][0], br[t][1], br[t][2], br[t][3],
                 br[t][4], br[t][5], br[t][6], br[t][7]};
      acc = wmma16(Wf[t][0], zf0, acc);
      acc = wmma16(Wf[t][1], zf1, acc);
#pragma unroll
      for (int j = 0; j < 4; ++j) {
        float h0 = fast_tanh(acc[2 * j]);
        float h1 = fast_tanh(acc[2 * j + 1]);
        g = fmaf(1.0f - h0 * h0, wr[t][2 * j], g);
        g = fmaf(1.0f - h1 * h1, wr[t][2 * j + 1], g);
        hp[t][j] = packh2(h0, h1);
      }
    }
    g += __shfl_xor(g, 16, 32);  // combine both w-halves of the trace
    if (lane < 16) dlp[mb + lane] = -g;

    // repack h into GEMM2 B-fragments: lane needs w in [16*hf,16*hf+16)+32s;
    // the missing 8 w-values live in the partner lane (l^16). One lane-half
    // exchange per packed pair yields both fragment halves.
    union BU { v16h v; float f[8]; } b0, b1;
#pragma unroll
    for (int j = 0; j < 4; ++j) {
      half_swap(hp[0][j], hp[1][j], &b0.f[j], &b0.f[4 + j]);
      half_swap(hp[2][j], hp[3][j], &b1.f[j], &b1.f[4 + j]);
    }

    // GEMM2: dz^T tiles; D element -> (d = t*16 + i + 8*hf, batch = mb+r)
#pragma unroll
    for (int t = 0; t < 4; ++t) {
      v8f dacc = {0.f, 0.f, 0.f, 0.f, 0.f, 0.f, 0.f, 0.f};
      dacc = wmma16(Uf[t][0], b0.v, dacc);
      dacc = wmma16(Uf[t][1], b1.v, dacc);
      float* ob = dz + (size_t)(mb + r) * 64 + t * 16 + 8 * hf;
      fx4 s0, s1;
#pragma unroll
      for (int i = 0; i < 4; ++i) { s0[i] = dacc[i]; s1[i] = dacc[4 + i]; }
      *(fx4*)ob       = s0;   // merged into b128 store clauses
      *(fx4*)(ob + 4) = s1;
    }
  }
}

// ---------------------------------------------------------------------------
extern "C" void kernel_launch(void* const* d_in, const int* in_sizes, int n_in,
                              void* d_out, int out_size, void* d_ws,
                              size_t ws_size, hipStream_t stream) {
  (void)in_sizes; (void)n_in; (void)out_size; (void)ws_size;
  const float* t  = (const float*)d_in[0];
  const float* z  = (const float*)d_in[1];
  // d_in[2] = logp_z (unused by the math)
  const float* W1 = (const float*)d_in[3];
  const float* b1 = (const float*)d_in[4];
  const float* W2 = (const float*)d_in[5];
  const float* b2 = (const float*)d_in[6];
  const float* W3 = (const float*)d_in[7];
  const float* b3 = (const float*)d_in[8];

  char* ws = (char*)d_ws;
  float*    p2   = (float*)(ws + WS_P2);
  float*    p3   = (float*)(ws + WS_P3);
  _Float16* Wh   = (_Float16*)(ws + WS_WH);
  _Float16* Uth  = (_Float16*)(ws + WS_UTH);
  float*    wu   = (float*)(ws + WS_WU);
  float*    bias = (float*)(ws + WS_BIAS);

  float* out = (float*)d_out;
  float* dz  = out;                             // (BATCH, 64) flat
  float* dlp = out + (size_t)BATCH_N * D_DIM;   // (BATCH, 1) flat

  cnf_hyper1<<<1, 512, 0, stream>>>(t, W1, b1, W2, b2, p2);
  cnf_hyper2<<<(P3N + 127) / 128, 128, 0, stream>>>(p2, W3, b3, p3);
  cnf_derive<<<64, 64, 0, stream>>>(p3, Wh, Uth, wu, bias);
  cnf_main<<<1024, 256, 0, stream>>>(z, Wh, Uth, wu, bias, dz, dlp);
}
